// Self_Attn_58428735094786
// MI455X (gfx1250) — matrix operand order repaired and checked by reading, not compiled, not run on hardware
//
#include <hip/hip_runtime.h>

// ---------------------------------------------------------------------------
// MI455X (gfx1250) fused self-attention.
//
// Roofline: attention output (4x4096x4096 f32 = 268 MB) dominates -> ~11.5us
// floor at 23.3 TB/s. FLOPs (4.3 GF QK^T + 34.4 GF attn*V) are negligible for
// the WMMA units, so the kernel is organized to touch HBM minimally:
//   - q/k projections stored once as f16 in WMMA-friendly [B][W][32] layout
//   - one workgroup owns a 16-row energy stripe [16][4096] f32 in LDS
//     (256 KB -- only possible because CDNA5 gives 320 KB LDS per WGP),
//     does exact softmax in LDS, streams normalized attention to HBM once,
//     and reuses the LDS rows directly as WMMA A-fragments for attn*V.
// Matrix ops: v_wmma_f32_16x16x32_f16 (f16 in, f32 accumulate).
// This revision: attn*V loop builds each A-fragment once and feeds TWO
// accumulators (two c-blocks per wave), and all f32->f16 fragment builds use
// packed v_cvt_pk_rtz_f16_f32, halving non-WMMA VALU work in the hot loop.
// ---------------------------------------------------------------------------

typedef __attribute__((ext_vector_type(16))) _Float16 v16h;
typedef __attribute__((ext_vector_type(8)))  _Float16 v8h;
typedef __attribute__((ext_vector_type(8)))  float    v8f;

// __builtin_amdgcn_cvt_pkrtz returns a 2x __fp16 ext-vector; use its exact
// type for the union member so packed converts assign without a cast. The
// union reinterprets the same 256 bits as the v16h WMMA operand.
typedef decltype(__builtin_amdgcn_cvt_pkrtz(0.0f, 0.0f)) v2hpk;

#define BATCH 4
#define CDIM  256
#define WDIM  4096
#define DQK   32
#define QSCALE 0.0625f   // 1/sqrt(256)

union HFrag { v16h v; v2hpk h2[8]; };

// ---------------------------------------------------------------------------
// Kernel 1: q = scale*(Wq x0 + bq), k = Wk x1 + bk, stored f16 as [B][W][DQK].
// ~0.5 GFLOP total; x-loads are coalesced across n, weight row is scalar.
// ---------------------------------------------------------------------------
__global__ __launch_bounds__(256) void proj_qk(
    const float* __restrict__ x0, const float* __restrict__ x1,
    const float* __restrict__ wq, const float* __restrict__ bq,
    const float* __restrict__ wk, const float* __restrict__ bk,
    _Float16* __restrict__ qt, _Float16* __restrict__ kt)
{
    const int n  = blockIdx.x * 256 + threadIdx.x;   // position along W
    const int yy = blockIdx.y;                        // 0..63: [0,32)=q, [32,64)=k
    const int b  = blockIdx.z;
    const bool isK = (yy >= DQK);
    const int  d   = isK ? (yy - DQK) : yy;

    const float* __restrict__ x    = isK ? x1 : x0;
    const float* __restrict__ wrow = (isK ? wk : wq) + d * CDIM;
    const float bias = isK ? bk[d] : bq[d];

    const float* xp = x + (size_t)b * CDIM * WDIM + n;
    float acc = 0.f;
#pragma unroll 8
    for (int c = 0; c < CDIM; ++c)
        acc = fmaf(wrow[c], xp[(size_t)c * WDIM], acc);

    float v = acc + bias;
    if (!isK) v *= QSCALE;                 // fold softmax scale into q
    (isK ? kt : qt)[((size_t)b * WDIM + n) * DQK + d] = (_Float16)v;
}

// ---------------------------------------------------------------------------
// Kernel 2: fused energy -> softmax -> attention store -> attn*V (+ residual).
// One workgroup (256 thr = 8 waves) per (batch, 16-row m-block): 1024 blocks.
// Dynamic LDS: energy stripe [16][4096] f32 + reduction scratch (~257 KB).
// ---------------------------------------------------------------------------
__global__ __launch_bounds__(256) void attn_fused(
    const float* __restrict__ x0, const float* __restrict__ x1,
    const _Float16* __restrict__ qt, const _Float16* __restrict__ kt,
    const float* __restrict__ gamma,
    float* __restrict__ out, float* __restrict__ att)
{
    extern __shared__ float smem[];
    float* eLDS   = smem;                 // [16][WDIM] energy, then exp values
    float* red    = eLDS + 16 * WDIM;     // [16][16] partial reductions
    float* rowmax = red + 256;            // [16]
    float* rowinv = rowmax + 16;          // [16]

    const int tid  = threadIdx.x;
    const int wave = tid >> 5;
    const int lane = tid & 31;
    const int lg   = lane >> 4;           // half-wave group (K-range select)
    const int ln   = lane & 15;           // row/col within fragment

    const int b  = blockIdx.x >> 8;       // 256 m-blocks per batch
    const int m0 = (blockIdx.x & 255) * 16;

    // ---- A fragment for QK^T: k rows m0..m0+15, all 32 d (held all phase 1).
    // A 16x32 k order used here (contiguous): half i -> K = i + 16*lg (matches B below).
    // In [W][32] storage that is two contiguous 8-half (16B) chunks.
    const _Float16* kp = kt + ((size_t)b * WDIM + m0 + ln) * DQK;
    v8h k0 = *(const v8h*)(kp + lg * 16);
    v8h k1 = *(const v8h*)(kp + lg * 16 + 8);
    v16h aK;
#pragma unroll
    for (int i = 0; i < 8; ++i) { aK[i] = k0[i]; aK[8 + i] = k1[i]; }

    // ---- Phase 1: energy[16][4096] = k_block^T q  (256 WMMA tiles / block).
    // B 32x16 k order used here (same as A): half i -> K = i + 16*lg, N = ln  => one contiguous
    // 32B read per lane from qt[b][n0+ln][16*lg ..].
    for (int nt = wave; nt < WDIM / 16; nt += 8) {
        const int n0 = nt * 16;
        v16h bQ = *(const v16h*)(qt + ((size_t)b * WDIM + n0 + ln) * DQK + lg * 16);
        v8f e = {};
        e = __builtin_amdgcn_wmma_f32_16x16x32_f16(false, aK, false, bQ,
                                                   (short)0, e, false, false);
#pragma unroll
        for (int r = 0; r < 8; ++r)        // C/D layout: M=(lg?8:0)+r, N=ln
            eLDS[(size_t)((lg ? 8 : 0) + r) * WDIM + n0 + ln] = e[r];
    }
    __syncthreads();

    // ---- Phase 2: exact row softmax over the LDS stripe (16 thr per row).
    const int row = tid >> 4;
    const int j   = tid & 15;
    {
        float mx = -3.4e38f;
        for (int idx = j; idx < WDIM; idx += 16)
            mx = fmaxf(mx, eLDS[(size_t)row * WDIM + idx]);
        red[row * 16 + j] = mx;
    }
    __syncthreads();
    if (j == 0) {
        float mx = red[row * 16];
#pragma unroll
        for (int t = 1; t < 16; ++t) mx = fmaxf(mx, red[row * 16 + t]);
        rowmax[row] = mx;
    }
    __syncthreads();
    {
        const float rm = rowmax[row];
        float s = 0.f;
        for (int idx = j; idx < WDIM; idx += 16) {
            float v = __expf(eLDS[(size_t)row * WDIM + idx] - rm);
            eLDS[(size_t)row * WDIM + idx] = v;   // keep exp values in LDS
            s += v;
        }
        red[row * 16 + j] = s;
    }
    __syncthreads();
    if (j == 0) {
        float s = 0.f;
#pragma unroll
        for (int t = 0; t < 16; ++t) s += red[row * 16 + t];
        rowinv[row] = 1.f / s;
    }
    __syncthreads();

    // ---- Phase 3a: stream normalized attention to HBM (written exactly once,
    // fully coalesced: consecutive tid -> consecutive n).
    for (int idx = tid; idx < 16 * WDIM; idx += 256) {
        const int r = idx >> 12;           // WDIM = 2^12
        att[((size_t)b * WDIM + m0 + r) * WDIM + (idx & (WDIM - 1))]
            = eLDS[idx] * rowinv[r];
    }

    // ---- Phase 3b: out_block[16 m][256 c] = attn * x0^T, K=4096 from LDS.
    // Each wave owns two 16-wide c-blocks (c0 and c0+128); the A-fragment is
    // built ONCE per K-tile and feeds two accumulators back-to-back.
    const float g0   = gamma[0];
    const float invA = rowinv[ln];         // A-fragment row M = ln
    const int   c0   = wave * 16;
    const float* xrow0 = x0 + ((size_t)b * CDIM + c0 + ln) * WDIM + lg * 16;
    const float* xrow1 = xrow0 + (size_t)128 * WDIM;   // second c-block
    v8f acc0 = {}, acc1 = {};

    for (int nt = 0; nt < WDIM / 32; ++nt) {
        // A: exp rows from LDS, K=16lg..16lg+15 contiguous (b128 reads),
        // 1/rowsum folded in during the packed f32->f16 convert.
        const float4* ap =
            (const float4*)(eLDS + (size_t)ln * WDIM + nt * 32 + lg * 16);
        float4 a0 = ap[0], a1 = ap[1], a2 = ap[2], a3 = ap[3];
        HFrag A;
        A.h2[0] = __builtin_amdgcn_cvt_pkrtz(a0.x * invA, a0.y * invA);
        A.h2[1] = __builtin_amdgcn_cvt_pkrtz(a0.z * invA, a0.w * invA);
        A.h2[2] = __builtin_amdgcn_cvt_pkrtz(a1.x * invA, a1.y * invA);
        A.h2[3] = __builtin_amdgcn_cvt_pkrtz(a1.z * invA, a1.w * invA);
        A.h2[4] = __builtin_amdgcn_cvt_pkrtz(a2.x * invA, a2.y * invA);
        A.h2[5] = __builtin_amdgcn_cvt_pkrtz(a2.z * invA, a2.w * invA);
        A.h2[6] = __builtin_amdgcn_cvt_pkrtz(a3.x * invA, a3.y * invA);
        A.h2[7] = __builtin_amdgcn_cvt_pkrtz(a3.z * invA, a3.w * invA);

        // B tiles for both c-blocks: x0, K contiguous (4x float4 per lane).
        const float4* bp0 = (const float4*)(xrow0 + nt * 32);
        const float4* bp1 = (const float4*)(xrow1 + nt * 32);
        if (nt + 1 < WDIM / 32) {
            __builtin_prefetch(xrow0 + (nt + 1) * 32, 0, 1);
            __builtin_prefetch(xrow1 + (nt + 1) * 32, 0, 1);
        }
        float4 b0 = bp0[0], b1 = bp0[1], b2 = bp0[2], b3 = bp0[3];
        float4 c4 = bp1[0], c5 = bp1[1], c6 = bp1[2], c7 = bp1[3];
        HFrag B0, B1;
        B0.h2[0] = __builtin_amdgcn_cvt_pkrtz(b0.x, b0.y);
        B0.h2[1] = __builtin_amdgcn_cvt_pkrtz(b0.z, b0.w);
        B0.h2[2] = __builtin_amdgcn_cvt_pkrtz(b1.x, b1.y);
        B0.h2[3] = __builtin_amdgcn_cvt_pkrtz(b1.z, b1.w);
        B0.h2[4] = __builtin_amdgcn_cvt_pkrtz(b2.x, b2.y);
        B0.h2[5] = __builtin_amdgcn_cvt_pkrtz(b2.z, b2.w);
        B0.h2[6] = __builtin_amdgcn_cvt_pkrtz(b3.x, b3.y);
        B0.h2[7] = __builtin_amdgcn_cvt_pkrtz(b3.z, b3.w);
        B1.h2[0] = __builtin_amdgcn_cvt_pkrtz(c4.x, c4.y);
        B1.h2[1] = __builtin_amdgcn_cvt_pkrtz(c4.z, c4.w);
        B1.h2[2] = __builtin_amdgcn_cvt_pkrtz(c5.x, c5.y);
        B1.h2[3] = __builtin_amdgcn_cvt_pkrtz(c5.z, c5.w);
        B1.h2[4] = __builtin_amdgcn_cvt_pkrtz(c6.x, c6.y);
        B1.h2[5] = __builtin_amdgcn_cvt_pkrtz(c6.z, c6.w);
        B1.h2[6] = __builtin_amdgcn_cvt_pkrtz(c7.x, c7.y);
        B1.h2[7] = __builtin_amdgcn_cvt_pkrtz(c7.z, c7.w);

        acc0 = __builtin_amdgcn_wmma_f32_16x16x32_f16(false, A.v, false, B0.v,
                                                      (short)0, acc0, false, false);
        acc1 = __builtin_amdgcn_wmma_f32_16x16x32_f16(false, A.v, false, B1.v,
                                                      (short)0, acc1, false, false);
    }

    // out[b][c][m] = gamma*acc + x1[b][c][m]   (C/D: M=(lg?8:0)+r, N=ln)
#pragma unroll
    for (int r = 0; r < 8; ++r) {
        const size_t o0 = ((size_t)b * CDIM + c0 + ln) * WDIM
                        + m0 + (lg ? 8 : 0) + r;
        const size_t o1 = o0 + (size_t)128 * WDIM;
        out[o0] = g0 * acc0[r] + x1[o0];
        out[o1] = g0 * acc1[r] + x1[o1];
    }
}

// ---------------------------------------------------------------------------
extern "C" void kernel_launch(void* const* d_in, const int* in_sizes, int n_in,
                              void* d_out, int out_size, void* d_ws, size_t ws_size,
                              hipStream_t stream)
{
    const float* x0    = (const float*)d_in[0];
    const float* x1    = (const float*)d_in[1];
    const float* wq    = (const float*)d_in[2];
    const float* bq    = (const float*)d_in[3];
    const float* wk    = (const float*)d_in[4];
    const float* bk    = (const float*)d_in[5];
    const float* gamma = (const float*)d_in[6];

    _Float16* qt = (_Float16*)d_ws;                         // 1 MB
    _Float16* kt = qt + (size_t)BATCH * WDIM * DQK;         // 1 MB

    float* out = (float*)d_out;                             // [B][C][W]
    float* att = out + (size_t)BATCH * CDIM * WDIM;         // [B][W][W]

    proj_qk<<<dim3(WDIM / 256, 2 * DQK, BATCH), 256, 0, stream>>>(
        x0, x1, wq, bq, wk, bk, qt, kt);

    const size_t ldsBytes = (size_t)(16 * WDIM + 256 + 16 + 16) * sizeof(float);
    attn_fused<<<dim3(BATCH * (WDIM / 16)), 256, ldsBytes, stream>>>(
        x0, x1, qt, kt, gamma, out, att);
}
